// SelfAttention_50646254355056
// MI455X (gfx1250) — matrix operand order repaired and checked by reading, not compiled, not run on hardware
//
#include <hip/hip_runtime.h>

// ---------------------------------------------------------------------------
// Flash-attention forward for gfx1250 (MI455X), wave32 + v_wmma_f32_16x16x32_f16
// Reference: B=2, S=4096, H=8, D=64; qkv (B,S,3,H,D) fp32; gist mask (B,S) bool;
// query_start scalar int. Output (B,S,H,D) fp32.
// ---------------------------------------------------------------------------

typedef _Float16 v16h __attribute__((ext_vector_type(16)));
typedef _Float16 v8h  __attribute__((ext_vector_type(8)));
typedef float    v8f  __attribute__((ext_vector_type(8)));

union V16U { v16h v; v8h h[2]; };

constexpr int Bc = 2, Sc = 4096, Hc = 8, Dc = 64;
constexpr int BQ = 128;                 // queries per workgroup (8 waves x 16)
constexpr int KT = 32;                  // keys per iteration
constexpr int QKV_ROW = 3 * Hc * Dc;    // 1536 floats between seq positions

// reduce across the 16 lanes of each wave half (lanes 0-15 / 16-31 independent)
__device__ __forceinline__ float rowmax16(float v) {
  v = fmaxf(v, __shfl_xor(v, 1, 32));
  v = fmaxf(v, __shfl_xor(v, 2, 32));
  v = fmaxf(v, __shfl_xor(v, 4, 32));
  v = fmaxf(v, __shfl_xor(v, 8, 32));
  return v;
}
__device__ __forceinline__ float rowsum16(float v) {
  v += __shfl_xor(v, 1, 32);
  v += __shfl_xor(v, 2, 32);
  v += __shfl_xor(v, 4, 32);
  v += __shfl_xor(v, 8, 32);
  return v;
}

__device__ __forceinline__ v8h cvt8(const float4 a, const float4 b) {
  v8h r;
  r[0] = (_Float16)a.x; r[1] = (_Float16)a.y; r[2] = (_Float16)a.z; r[3] = (_Float16)a.w;
  r[4] = (_Float16)b.x; r[5] = (_Float16)b.y; r[6] = (_Float16)b.z; r[7] = (_Float16)b.w;
  return r;
}

__global__ __launch_bounds__(256)
void fa_fwd_gfx1250(const float* __restrict__ qkv,
                    const unsigned char* __restrict__ gist,
                    const int* __restrict__ qstart_p,
                    float* __restrict__ out)
{
  // K tile: [32 keys][64 dims] f16, rows padded to 72 halves (144B, 16B-mult)
  __shared__ _Float16 sK[32 * 72];
  // V tile transposed: [64 dims][32 keys] f16, rows padded to 40 halves (80B)
  __shared__ _Float16 sV[64 * 40];
  // per-wave P scratch: [8 waves][16 rows][32 keys] padded to 40 halves/row
  __shared__ _Float16 sP[8 * 16 * 40];

  const int tid  = threadIdx.x;
  const int wave = tid >> 5, lane = tid & 31;
  const int lh   = lane >> 4, ln = lane & 15;       // half id, lane-in-half
  const int b    = blockIdx.y >> 3, h = blockIdx.y & 7;
  const int qBase  = blockIdx.x * BQ;
  const int qstart = qstart_p[0];
  const int tRow   = qBase + wave * 16;             // first query row of this wave

  const float* qB = qkv + (size_t)b * Sc * QKV_ROW + 0 * Hc * Dc + h * Dc;
  const float* kB = qkv + (size_t)b * Sc * QKV_ROW + 1 * Hc * Dc + h * Dc;
  const float* vB = qkv + (size_t)b * Sc * QKV_ROW + 2 * Hc * Dc + h * Dc;
  const unsigned char* gb = gist + (size_t)b * Sc;

  // ---- Q tile -> two 16x32 A-fragments (dims 0-31, 32-63), fp16 in registers.
  // A fragment as loaded here (contiguous k, 16x32): element i of lane l holds
  //   K = (l/16)*16 + i, row M = l%16.
  v16h aq[2];
  {
    const float* qrow = qB + (size_t)(tRow + ln) * QKV_ROW;
#pragma unroll
    for (int s = 0; s < 2; ++s) {
#pragma unroll
      for (int g = 0; g < 2; ++g) {
        const int d0 = s * 32 + lh * 16 + g * 8;    // 8 consecutive dims
        float4 f0 = *reinterpret_cast<const float4*>(qrow + d0);
        float4 f1 = *reinterpret_cast<const float4*>(qrow + d0 + 4);
        v8h c = cvt8(f0, f1);
#pragma unroll
        for (int i = 0; i < 8; ++i) aq[s][g * 8 + i] = c[i];
      }
    }
  }

  // C/D layout: lane l holds column N = l%16, rows M = r + 8*(l/16), r=0..7.
  v8f acc[4];                 // 16x64 output accumulator (4 N-tiles of 16 dims)
  float m_i[8], l_i[8];       // per-row running max / sum (replicated per half)
#pragma unroll
  for (int r = 0; r < 8; ++r) { m_i[r] = -1e30f; l_i[r] = 0.f; }
#pragma unroll
  for (int d = 0; d < 4; ++d)
#pragma unroll
    for (int r = 0; r < 8; ++r) acc[d][r] = 0.f;

  // cooperative-load work assignments
  const int kKey = tid >> 3, kDim = (tid & 7) * 8;  // 32 keys x 8-dim chunks
  const int vDim = tid & 63, vKey = (tid >> 6) * 8; // 64 dims x 8-key chunks
  const int nIter = (qBase + BQ) / KT;              // causal: keys 0..qBase+127

  for (int it = 0; it < nIter; ++it) {
    const int j = it * KT;
    __syncthreads();  // previous iteration's fragment reads must finish

    { // K tile -> sK[key][dim] as f16
      const float* src = kB + (size_t)(j + kKey) * QKV_ROW + kDim;
      float4 f0 = *reinterpret_cast<const float4*>(src);
      float4 f1 = *reinterpret_cast<const float4*>(src + 4);
      *reinterpret_cast<v8h*>(&sK[kKey * 72 + kDim]) = cvt8(f0, f1);
    }
    { // V tile -> sV[dim][key] (transposed) as f16
      const float* src = vB + (size_t)(j + vKey) * QKV_ROW + vDim;
      v8h hv;
#pragma unroll
      for (int kk = 0; kk < 8; ++kk) hv[kk] = (_Float16)src[(size_t)kk * QKV_ROW];
      *reinterpret_cast<v8h*>(&sV[vDim * 40 + vKey]) = hv;
    }
    __syncthreads();

    // ---- S = Q K^T for two 16-key subtiles (B layout: lane l = column l%16,
    //      element i = K-dim (l/16)*16 + i within the 32-wide slice).
    v8f sc[2];
#pragma unroll
    for (int sub = 0; sub < 2; ++sub) {
      const _Float16* kr = &sK[(sub * 16 + ln) * 72 + lh * 16];
      V16U b0, b1;
      b0.h[0] = *reinterpret_cast<const v8h*>(kr);
      b0.h[1] = *reinterpret_cast<const v8h*>(kr + 8);
      b1.h[0] = *reinterpret_cast<const v8h*>(kr + 32);   // dims 32..63 slice
      b1.h[1] = *reinterpret_cast<const v8h*>(kr + 40);
      v8f c = {};
      c = __builtin_amdgcn_wmma_f32_16x16x32_f16(false, aq[0], false, b0.v,
                                                 (short)0, c, false, false);
      c = __builtin_amdgcn_wmma_f32_16x16x32_f16(false, aq[1], false, b1.v,
                                                 (short)0, c, false, false);
      sc[sub] = c;
    }

    // ---- scale + additive mask (matches reference: -10000, not -inf)
    const int s0 = j + ln, s1 = j + 16 + ln;
    const bool v0 = (gb[s0] != 0) || (s0 >= qstart);
    const bool v1 = (gb[s1] != 0) || (s1 >= qstart);
#pragma unroll
    for (int r = 0; r < 8; ++r) {
      const int t = tRow + lh * 8 + r;
      const bool rf = (t < qstart);
      sc[0][r] = sc[0][r] * 0.125f + (((s0 <= t) && (v0 || rf)) ? 0.f : -10000.f);
      sc[1][r] = sc[1][r] * 0.125f + (((s1 <= t) && (v1 || rf)) ? 0.f : -10000.f);
    }

    // ---- online softmax; write P (f16) into per-wave LDS scratch
    _Float16* pw = &sP[wave * 640];
#pragma unroll
    for (int r = 0; r < 8; ++r) {
      const float tmax  = rowmax16(fmaxf(sc[0][r], sc[1][r]));
      const float mnew  = fmaxf(m_i[r], tmax);
      const float alpha = __expf(m_i[r] - mnew);
      const float p0 = __expf(sc[0][r] - mnew);
      const float p1 = __expf(sc[1][r] - mnew);
      l_i[r] = l_i[r] * alpha + rowsum16(p0 + p1);
      m_i[r] = mnew;
#pragma unroll
      for (int d = 0; d < 4; ++d) acc[d][r] *= alpha;
      const int mr = lh * 8 + r;
      pw[mr * 40 + ln]      = (_Float16)p0;
      pw[mr * 40 + 16 + ln] = (_Float16)p1;
    }
    // cross-lane C-layout -> A-layout transpose through LDS: fence DS ops
    asm volatile("s_wait_dscnt 0" ::: "memory");

    // ---- P as 16x32 A-fragment
    const _Float16* pr = &sP[wave * 640 + ln * 40];
    V16U ap;
    ap.h[0] = *reinterpret_cast<const v8h*>(pr + lh * 16);
    ap.h[1] = *reinterpret_cast<const v8h*>(pr + lh * 16 + 8);

    // ---- O += P V  (V B-fragments contiguous thanks to transposed sV)
#pragma unroll
    for (int d = 0; d < 4; ++d) {
      const _Float16* vr = &sV[(d * 16 + ln) * 40 + lh * 16];
      V16U bv;
      bv.h[0] = *reinterpret_cast<const v8h*>(vr);
      bv.h[1] = *reinterpret_cast<const v8h*>(vr + 8);
      acc[d] = __builtin_amdgcn_wmma_f32_16x16x32_f16(false, ap.v, false, bv.v,
                                                      (short)0, acc[d], false, false);
    }
  }

  // ---- epilogue: normalize rows, store fp32 (b, t, h, d)
  float* ob = out + ((size_t)b * Sc + tRow) * Hc * Dc + h * Dc;
#pragma unroll
  for (int r = 0; r < 8; ++r) {
    const int mr = lh * 8 + r;
    const float inv = 1.0f / l_i[r];
    float* orow = ob + (size_t)mr * Hc * Dc;
#pragma unroll
    for (int d = 0; d < 4; ++d) orow[d * 16 + ln] = acc[d][r] * inv;
  }
}

extern "C" void kernel_launch(void* const* d_in, const int* in_sizes, int n_in,
                              void* d_out, int out_size, void* d_ws, size_t ws_size,
                              hipStream_t stream) {
  (void)in_sizes; (void)n_in; (void)out_size; (void)d_ws; (void)ws_size;
  const float* qkv          = (const float*)d_in[0];
  const unsigned char* gist = (const unsigned char*)d_in[1];  // bool mask, 1B/elem
  const int* qstart         = (const int*)d_in[2];
  float* out                = (float*)d_out;

  dim3 grid(Sc / BQ, Bc * Hc);   // (32 query blocks, 16 batch*head)
  fa_fwd_gfx1250<<<grid, dim3(256), 0, stream>>>(qkv, gist, qstart, out);
}
